// HybridAttention_27642409517747
// MI455X (gfx1250) — matrix operand order repaired and checked by reading, not compiled, not run on hardware
//
#include <hip/hip_runtime.h>
#include <hip/hip_bf16.h>
#include <cstdint>
#include <cstddef>

// ---------------- problem constants (layer_idx = 1 as set up by the harness) ----
#define NSEQ    2048
#define DMODEL  1024
#define NHEADS  16
#define HDIM    64
#define BATCH   2
#define PLO     478          // band start  (layer_idx=1 -> p=478, q=785)
#define NFREQ   307          // q - p
#define NFPAD   640          // 2*NFREQ=614 padded up to multiple of 32
#define TOPKK   15           // int(2*ln(2049)) = 15
#define GAMMA_W 0.5f

typedef __bf16 bf16;
typedef __attribute__((ext_vector_type(16))) __bf16 v16bf;
typedef __attribute__((ext_vector_type(8)))  __bf16 v8bf;
typedef __attribute__((ext_vector_type(8)))  float  v8f;

union FragA { v16bf v; v8bf h[2]; };

// A-matrix 16x32 bf16 fragment (contiguous k order, same as load_b_frag):
// lane<16 : row=lane,    K = k0+0..15
// lane>=16: row=lane-16, K = k0+16..31
__device__ __forceinline__ v16bf load_a_frag(const bf16* base, int row0, int ld, int k0) {
  int l  = threadIdx.x & 31;
  int r  = l & 15, hh = l >> 4;
  const bf16* p = base + (size_t)(row0 + r) * ld + k0 + hh * 16;
  FragA f;
  f.h[0] = *(const v8bf*)(p);
  f.h[1] = *(const v8bf*)(p + 8);
  return f.v;
}

// B-matrix 32x16 bf16 fragment: lane<16 col=lane K=k0..k0+15 ; lane>=16 col=lane-16 K=k0+16..31
// Memory must be column-contiguous (i.e. B^T row-major with stride ld).
__device__ __forceinline__ v16bf load_b_frag(const bf16* base, int col0, int ld, int k0) {
  int l  = threadIdx.x & 31;
  int cc = l & 15, hh = l >> 4;
  return *(const v16bf*)(base + (size_t)(col0 + cc) * ld + k0 + hh * 16);
}

__device__ __forceinline__ v8f wmma_bf16(v16bf a, v16bf b, v8f c) {
  return __builtin_amdgcn_wmma_f32_16x16x32_bf16(false, a, false, b, (short)0, c, false, false);
}

// ---------------------------------------------------------------- fp32 -> bf16 --
__global__ void k_cvt_bf16(const float* __restrict__ src, bf16* __restrict__ dst, int n) {
  int i = blockIdx.x * blockDim.x + threadIdx.x;
  if (i < n) dst[i] = (bf16)src[i];
}

// -------------------------------------------------- DFT basis: F[640][2048], G[2048][640]
// F rows 0..306  : cos(2*pi*(PLO+f)*n/N)      (Re extractor)
// F rows 307..613: sin(2*pi*(PLO+f)*n/N)      (S = -Im extractor)
// rows >=614 zero.  G = (2/N) * F^T (zero padded cols).
__global__ void k_init_basis(bf16* __restrict__ F, bf16* __restrict__ G) {
  int idx = blockIdx.x * 256 + threadIdx.x;
  if (idx >= NSEQ * NFPAD) return;
  int n = idx / NFPAD;
  int f = idx % NFPAD;
  float fv = 0.0f, gv = 0.0f;
  if (f < 2 * NFREQ) {
    int fr = (f < NFREQ) ? f : f - NFREQ;
    int m  = PLO + fr;
    int phase = (int)(((long long)m * (long long)n) & (NSEQ - 1)); // mod 2048
    float th = (float)phase * (6.283185307179586f / (float)NSEQ);
    float s, c;
    sincosf(th, &s, &c);
    float v = (f < NFREQ) ? c : s;
    fv = v;
    gv = v * (2.0f / (float)NSEQ);
  }
  F[(size_t)f * NSEQ + n] = (bf16)fv;
  G[(size_t)n * NFPAD + f] = (bf16)gv;
}

// ------------------------------------------------------------ generic WMMA GEMM --
// C[M,N] = A[M,K] (row-major bf16) * B (column-contiguous bf16, stride ldb) + bias
// Wave tile 64x64 (4x4 WMMA tiles), block = 4 waves -> 128x128 per block.
// Outputs (any subset): bf16 row-major, bf16 transposed (with per-chunk batching), fp32 row-major.
__global__ __launch_bounds__(128) void k_wmma_gemm(
    const bf16* __restrict__ A,  size_t sAb,
    const bf16* __restrict__ Bc, size_t sBb,
    int M, int N, int K, int lda, int ldb,
    const float* __restrict__ bias,
    bf16* Crm, size_t sCrm, int ldc,
    bf16* Ct,  size_t sCt,  int ldt, int rpc, size_t chunkStride,
    float* C32, size_t sC32, int ldc32)
{
  int wave = threadIdx.x >> 5;
  int lane = threadIdx.x & 31;
  int cc = lane & 15, hh = lane >> 4;
  int mb = blockIdx.y * 128 + (wave >> 1) * 64;
  int nb = blockIdx.x * 128 + (wave & 1) * 64;
  const bf16* Ab = A  + (size_t)blockIdx.z * sAb;
  const bf16* Bb = Bc + (size_t)blockIdx.z * sBb;

  v8f zero = {0.f,0.f,0.f,0.f,0.f,0.f,0.f,0.f};
  v8f acc[4][4];
#pragma unroll
  for (int i = 0; i < 4; ++i)
#pragma unroll
    for (int j = 0; j < 4; ++j) acc[i][j] = zero;

  for (int kk = 0; kk < K; kk += 32) {
    v16bf a[4], bf[4];
#pragma unroll
    for (int i = 0; i < 4; ++i) a[i]  = load_a_frag(Ab, mb + i * 16, lda, kk);
#pragma unroll
    for (int j = 0; j < 4; ++j) bf[j] = load_b_frag(Bb, nb + j * 16, ldb, kk);
    if (kk + 64 < K) { // prefetch hints -> global_prefetch_b8
      __builtin_prefetch(Ab + (size_t)(mb + cc) * lda + kk + 64, 0, 1);
      __builtin_prefetch(Bb + (size_t)(nb + cc) * ldb + kk + 64, 0, 1);
    }
#pragma unroll
    for (int i = 0; i < 4; ++i)
#pragma unroll
      for (int j = 0; j < 4; ++j)
        acc[i][j] = wmma_bf16(a[i], bf[j], acc[i][j]);
  }

#pragma unroll
  for (int i = 0; i < 4; ++i) {
#pragma unroll
    for (int j = 0; j < 4; ++j) {
      int m0 = mb + i * 16, n0 = nb + j * 16;
      int n = n0 + cc;
      float bval = bias ? bias[n] : 0.0f;
#pragma unroll
      for (int e = 0; e < 8; ++e) {
        int m = m0 + e + 8 * hh;
        float val = acc[i][j][e] + bval;
        if (Crm) Crm[(size_t)blockIdx.z * sCrm + (size_t)m * ldc + n] = (bf16)val;
        if (Ct)  Ct [(size_t)blockIdx.z * sCt + (size_t)(m / rpc) * chunkStride
                     + (size_t)n * ldt + (m % rpc)] = (bf16)val;
        if (C32) C32[(size_t)blockIdx.z * sC32 + (size_t)m * ldc32 + n] = val;
      }
    }
  }
}

// ------------------------------------------------------- flash attention (WMMA) --
// Qb,Kb : bf16 row-major [B][N][D] (band-filtered).  VbT : bf16 [B][D][N] (transposed).
// One wave per (b, head, 16-query tile).  Online softmax, causal mask.
__global__ __launch_bounds__(32) void k_flash_attn(
    const bf16* __restrict__ Qb, const bf16* __restrict__ Kb,
    const bf16* __restrict__ VbT, float* __restrict__ timeO)
{
  int qb = blockIdx.x * 16;
  int h  = blockIdx.y;
  int b  = blockIdx.z;
  int lane = threadIdx.x & 31;
  int cc = lane & 15, hh = lane >> 4;

  const bf16* Qbase = Qb  + (size_t)b * NSEQ * DMODEL;
  const bf16* Kbase = Kb  + (size_t)b * NSEQ * DMODEL;
  const bf16* Vbase = VbT + (size_t)b * DMODEL * NSEQ;

  v16bf a0 = load_a_frag(Qbase, qb, DMODEL, h * HDIM + 0);
  v16bf a1 = load_a_frag(Qbase, qb, DMODEL, h * HDIM + 32);

  v8f zero = {0.f,0.f,0.f,0.f,0.f,0.f,0.f,0.f};
  v8f acc[4];
#pragma unroll
  for (int t = 0; t < 4; ++t) acc[t] = zero;

  float m_i[8], l_i[8];
#pragma unroll
  for (int e = 0; e < 8; ++e) { m_i[e] = -1e30f; l_i[e] = 0.0f; }

  __shared__ bf16 Plds[16 * 32];

  for (int kb = 0; kb < qb + 16; kb += 32) {
    v16bf b00 = load_b_frag(Kbase, kb,      DMODEL, h * HDIM + 0);
    v16bf b01 = load_b_frag(Kbase, kb,      DMODEL, h * HDIM + 32);
    v16bf b10 = load_b_frag(Kbase, kb + 16, DMODEL, h * HDIM + 0);
    v16bf b11 = load_b_frag(Kbase, kb + 16, DMODEL, h * HDIM + 32);

    v8f s0 = wmma_bf16(a1, b01, wmma_bf16(a0, b00, zero));
    v8f s1 = wmma_bf16(a1, b11, wmma_bf16(a0, b10, zero));

    float p0[8], p1[8], mnew[8], rs[8];
#pragma unroll
    for (int e = 0; e < 8; ++e) {
      int row = qb + e + 8 * hh;
      float v0 = s0[e] * 0.125f;                    // 1/sqrt(64)
      float v1 = s1[e] * 0.125f;
      if (kb + cc      > row) v0 = -1e30f;          // causal mask
      if (kb + 16 + cc > row) v1 = -1e30f;
      p0[e] = v0; p1[e] = v1;
      mnew[e] = fmaxf(v0, v1);
    }
#pragma unroll
    for (int off = 1; off < 16; off <<= 1)
#pragma unroll
      for (int e = 0; e < 8; ++e)
        mnew[e] = fmaxf(mnew[e], __shfl_xor(mnew[e], off, 32));
#pragma unroll
    for (int e = 0; e < 8; ++e) {
      mnew[e] = fmaxf(mnew[e], m_i[e]);
      float corr = __expf(m_i[e] - mnew[e]);
      p0[e] = __expf(p0[e] - mnew[e]);
      p1[e] = __expf(p1[e] - mnew[e]);
      rs[e] = p0[e] + p1[e];
      l_i[e] *= corr;
      m_i[e] = mnew[e];
#pragma unroll
      for (int t = 0; t < 4; ++t) acc[t][e] *= corr;
    }
#pragma unroll
    for (int off = 1; off < 16; off <<= 1)
#pragma unroll
      for (int e = 0; e < 8; ++e)
        rs[e] += __shfl_xor(rs[e], off, 32);
#pragma unroll
    for (int e = 0; e < 8; ++e) l_i[e] += rs[e];

    // stage P (C-layout) into LDS, re-read as A fragment
#pragma unroll
    for (int e = 0; e < 8; ++e) {
      int row = e + 8 * hh;
      Plds[row * 32 + cc]      = (bf16)p0[e];
      Plds[row * 32 + 16 + cc] = (bf16)p1[e];
    }
    asm volatile("s_wait_dscnt 0" ::: "memory");
    __syncthreads();
    v16bf pf = load_a_frag(Plds, 0, 32, 0);

#pragma unroll
    for (int t = 0; t < 4; ++t) {
      v16bf vb = load_b_frag(Vbase, h * HDIM + t * 16, NSEQ, kb);
      acc[t] = wmma_bf16(pf, vb, acc[t]);
    }
    __syncthreads();
  }

#pragma unroll
  for (int t = 0; t < 4; ++t)
#pragma unroll
    for (int e = 0; e < 8; ++e) {
      int row = qb + e + 8 * hh;
      int d   = h * HDIM + t * 16 + cc;
      timeO[((size_t)b * NSEQ + row) * DMODEL + d] = acc[t][e] / l_i[e];
    }
}

// ----------------------------------- corr over D:  cr/ci [B][307] from YTq,YTk --
__global__ void k_corr_reduce(const bf16* __restrict__ YTq, const bf16* __restrict__ YTk,
                              float* __restrict__ cr, float* __restrict__ ci)
{
  int f = blockIdx.x;             // 0..306
  int b = blockIdx.y;
  const bf16* q = YTq + (size_t)b * DMODEL * NFPAD;
  const bf16* k = YTk + (size_t)b * DMODEL * NFPAD;
  float re = 0.f, im = 0.f;
  for (int d = threadIdx.x; d < DMODEL; d += 256) {
    float qc = (float)q[(size_t)d * NFPAD + f];
    float qs = (float)q[(size_t)d * NFPAD + NFREQ + f];
    float kc = (float)k[(size_t)d * NFPAD + f];
    float ks = (float)k[(size_t)d * NFPAD + NFREQ + f];
    re += qc * kc + qs * ks;      // Re(Qf * conj(Kf))
    im += qc * ks - qs * kc;      // Im(Qf * conj(Kf))
  }
  __shared__ float sre[256], sim[256];
  sre[threadIdx.x] = re; sim[threadIdx.x] = im;
  __syncthreads();
  for (int s = 128; s > 0; s >>= 1) {
    if ((int)threadIdx.x < s) { sre[threadIdx.x] += sre[threadIdx.x + s];
                                sim[threadIdx.x] += sim[threadIdx.x + s]; }
    __syncthreads();
  }
  if (threadIdx.x == 0) { cr[b * NFREQ + f] = sre[0]; ci[b * NFREQ + f] = sim[0]; }
}

// ------------------------------------------- R_mean[b][n] = (2/(N*D)) sum_f ... --
__global__ void k_rmean(const float* __restrict__ cr, const float* __restrict__ ci,
                        float* __restrict__ Rmean)
{
  int idx = blockIdx.x * 256 + threadIdx.x;    // 0..4095
  int b = idx >> 11;
  int n = idx & (NSEQ - 1);
  float accv = 0.f;
  for (int f = 0; f < NFREQ; ++f) {
    int m = PLO + f;
    int phase = (int)(((long long)m * (long long)n) & (NSEQ - 1));
    float th = (float)phase * (6.283185307179586f / (float)NSEQ);
    float s, c;
    sincosf(th, &s, &c);
    accv += cr[b * NFREQ + f] * c - ci[b * NFREQ + f] * s;
  }
  Rmean[idx] = accv * (2.0f / ((float)NSEQ * (float)DMODEL));
}

// ------------------------------------------------- top-15 + softmax per batch ----
__global__ void k_topk(float* __restrict__ Rmean, float* __restrict__ wout, int* __restrict__ tauout)
{
  int b = blockIdx.x;
  float* R = Rmean + (size_t)b * NSEQ;
  __shared__ float sv[256];
  __shared__ int   si[256];
  __shared__ float topv[TOPKK];
  __shared__ int   topi[TOPKK];
  for (int r = 0; r < TOPKK; ++r) {
    float best = -3.0e38f; int bi = 0;
    for (int n = threadIdx.x; n < NSEQ; n += 256) {
      float v = R[n];
      if (v > best) { best = v; bi = n; }
    }
    sv[threadIdx.x] = best; si[threadIdx.x] = bi;
    __syncthreads();
    for (int s = 128; s > 0; s >>= 1) {
      if ((int)threadIdx.x < s && sv[threadIdx.x + s] > sv[threadIdx.x]) {
        sv[threadIdx.x] = sv[threadIdx.x + s]; si[threadIdx.x] = si[threadIdx.x + s];
      }
      __syncthreads();
    }
    if (threadIdx.x == 0) { topv[r] = sv[0]; topi[r] = si[0]; R[si[0]] = -3.0e38f; }
    __syncthreads();
  }
  if (threadIdx.x == 0) {
    float mx = topv[0];
    for (int r = 1; r < TOPKK; ++r) mx = fmaxf(mx, topv[r]);
    float e[TOPKK], s = 0.f;
    for (int r = 0; r < TOPKK; ++r) { e[r] = __expf(topv[r] - mx); s += e[r]; }
    for (int r = 0; r < TOPKK; ++r) { wout[b * TOPKK + r] = e[r] / s; tauout[b * TOPKK + r] = topi[r]; }
  }
}

// ------------------------- freq_out[b,n,d] = sum_k w_k * V[b, (n - tau_k) mod N, d]
__global__ void k_gather(const float* __restrict__ Vf, const float* __restrict__ w,
                         const int* __restrict__ tau, float* __restrict__ freqO)
{
  int b = blockIdx.x >> 11;
  int n = blockIdx.x & (NSEQ - 1);
  float wv[TOPKK]; int tv[TOPKK];
#pragma unroll
  for (int r = 0; r < TOPKK; ++r) { wv[r] = w[b * TOPKK + r]; tv[r] = tau[b * TOPKK + r]; }
  for (int d = threadIdx.x; d < DMODEL; d += 256) {
    float a = 0.f;
#pragma unroll
    for (int r = 0; r < TOPKK; ++r) {
      int srcn = (n - tv[r]) & (NSEQ - 1);
      a += wv[r] * Vf[((size_t)b * NSEQ + srcn) * DMODEL + d];
    }
    freqO[((size_t)b * NSEQ + n) * DMODEL + d] = a;
  }
}

// ------------------------------------- H = gamma*time + (1-gamma)*freq  -> bf16 --
__global__ void k_combine(const float* __restrict__ t, const float* __restrict__ f,
                          bf16* __restrict__ out, int n)
{
  int i = blockIdx.x * blockDim.x + threadIdx.x;
  if (i < n) out[i] = (bf16)(GAMMA_W * t[i] + (1.0f - GAMMA_W) * f[i]);
}

// ================================================================= launcher =====
extern "C" void kernel_launch(void* const* d_in, const int* in_sizes, int n_in,
                              void* d_out, int out_size, void* d_ws, size_t ws_size,
                              hipStream_t stream) {
  (void)in_sizes; (void)n_in; (void)out_size; (void)ws_size;
  const float* X  = (const float*)d_in[0];
  // d_in[1] = attention_mask (causal; recomputed analytically on device)
  const float* Wq = (const float*)d_in[2];
  const float* bq = (const float*)d_in[3];
  const float* Wk = (const float*)d_in[4];
  const float* bk = (const float*)d_in[5];
  const float* Wv = (const float*)d_in[6];
  const float* bv = (const float*)d_in[7];
  const float* Wo = (const float*)d_in[8];
  const float* bo = (const float*)d_in[9];
  float* Out = (float*)d_out;

  const int BN = BATCH * NSEQ;                 // 4096
  char* p = (char*)d_ws;
  auto alloc = [&](size_t bytes) -> char* {
    char* r = p; p += (bytes + 255) & ~(size_t)255; return r;
  };

  bf16* Xbf   = (bf16*)alloc((size_t)BN * DMODEL * 2);          // also reused as Hc
  bf16* Wqb   = (bf16*)alloc((size_t)DMODEL * DMODEL * 2);
  bf16* Wkb   = (bf16*)alloc((size_t)DMODEL * DMODEL * 2);
  bf16* Wvb   = (bf16*)alloc((size_t)DMODEL * DMODEL * 2);
  bf16* Wob   = (bf16*)alloc((size_t)DMODEL * DMODEL * 2);
  bf16* QT    = (bf16*)alloc((size_t)BATCH * DMODEL * NSEQ * 2);
  bf16* KT    = (bf16*)alloc((size_t)BATCH * DMODEL * NSEQ * 2);
  bf16* VT    = (bf16*)alloc((size_t)BATCH * DMODEL * NSEQ * 2);
  float* Vf   = (float*)alloc((size_t)BN * DMODEL * 4);
  bf16* Fb    = (bf16*)alloc((size_t)NFPAD * NSEQ * 2);
  bf16* Gb    = (bf16*)alloc((size_t)NSEQ * NFPAD * 2);
  bf16* YTq   = (bf16*)alloc((size_t)BATCH * DMODEL * NFPAD * 2);
  bf16* YTk   = (bf16*)alloc((size_t)BATCH * DMODEL * NFPAD * 2);
  bf16* YTv   = (bf16*)alloc((size_t)BATCH * DMODEL * NFPAD * 2);
  bf16* Qband = (bf16*)alloc((size_t)BN * DMODEL * 2);
  bf16* Kband = (bf16*)alloc((size_t)BN * DMODEL * 2);
  bf16* VbT   = (bf16*)alloc((size_t)BATCH * DMODEL * NSEQ * 2);
  float* timeO = (float*)alloc((size_t)BN * DMODEL * 4);
  float* freqO = (float*)alloc((size_t)BN * DMODEL * 4);
  float* cr    = (float*)alloc((size_t)BATCH * NFREQ * 4);
  float* ci    = (float*)alloc((size_t)BATCH * NFREQ * 4);
  float* Rmean = (float*)alloc((size_t)BATCH * NSEQ * 4);
  float* wbuf  = (float*)alloc((size_t)BATCH * TOPKK * 4);
  int*   taub  = (int*)alloc((size_t)BATCH * TOPKK * 4);
  bf16* Hc = Xbf;                               // reuse: X consumed after QKV GEMMs

  const int nX = BN * DMODEL, nW = DMODEL * DMODEL;
  k_cvt_bf16<<<(nX + 255) / 256, 256, 0, stream>>>(X, Xbf, nX);
  k_cvt_bf16<<<(nW + 255) / 256, 256, 0, stream>>>(Wq, Wqb, nW);
  k_cvt_bf16<<<(nW + 255) / 256, 256, 0, stream>>>(Wk, Wkb, nW);
  k_cvt_bf16<<<(nW + 255) / 256, 256, 0, stream>>>(Wv, Wvb, nW);
  k_cvt_bf16<<<(nW + 255) / 256, 256, 0, stream>>>(Wo, Wob, nW);
  k_init_basis<<<(NSEQ * NFPAD + 255) / 256, 256, 0, stream>>>(Fb, Gb);

  // 1) QKV projections: C = X @ W^T + b ; write transposed bf16 [B][D][N] (and V in fp32)
  size_t tchunk = (size_t)DMODEL * NSEQ;
  k_wmma_gemm<<<dim3(8, 32, 1), 128, 0, stream>>>(
      Xbf, 0, Wqb, 0, BN, DMODEL, DMODEL, DMODEL, DMODEL, bq,
      nullptr, 0, 0, QT, 0, NSEQ, NSEQ, tchunk, nullptr, 0, 0);
  k_wmma_gemm<<<dim3(8, 32, 1), 128, 0, stream>>>(
      Xbf, 0, Wkb, 0, BN, DMODEL, DMODEL, DMODEL, DMODEL, bk,
      nullptr, 0, 0, KT, 0, NSEQ, NSEQ, tchunk, nullptr, 0, 0);
  k_wmma_gemm<<<dim3(8, 32, 1), 128, 0, stream>>>(
      Xbf, 0, Wvb, 0, BN, DMODEL, DMODEL, DMODEL, DMODEL, bv,
      nullptr, 0, 0, VT, 0, NSEQ, NSEQ, tchunk, Vf, 0, DMODEL);

  // 2) forward DFT: Y = F @ X_b, write transposed YT [B][D][NFPAD]
  size_t sB2 = (size_t)DMODEL * NSEQ, sT2 = (size_t)DMODEL * NFPAD;
  k_wmma_gemm<<<dim3(8, 5, 2), 128, 0, stream>>>(
      Fb, 0, QT, sB2, NFPAD, DMODEL, NSEQ, NSEQ, NSEQ, nullptr,
      nullptr, 0, 0, YTq, sT2, NFPAD, NFPAD, 0, nullptr, 0, 0);
  k_wmma_gemm<<<dim3(8, 5, 2), 128, 0, stream>>>(
      Fb, 0, KT, sB2, NFPAD, DMODEL, NSEQ, NSEQ, NSEQ, nullptr,
      nullptr, 0, 0, YTk, sT2, NFPAD, NFPAD, 0, nullptr, 0, 0);
  k_wmma_gemm<<<dim3(8, 5, 2), 128, 0, stream>>>(
      Fb, 0, VT, sB2, NFPAD, DMODEL, NSEQ, NSEQ, NSEQ, nullptr,
      nullptr, 0, 0, YTv, sT2, NFPAD, NFPAD, 0, nullptr, 0, 0);

  // 3) inverse DFT (band-limited): band = G @ Y ; Q,K row-major, V transposed
  size_t sRM = (size_t)NSEQ * DMODEL;
  k_wmma_gemm<<<dim3(8, 16, 2), 128, 0, stream>>>(
      Gb, 0, YTq, sT2, NSEQ, DMODEL, NFPAD, NFPAD, NFPAD, nullptr,
      Qband, sRM, DMODEL, nullptr, 0, 0, 0, 0, nullptr, 0, 0);
  k_wmma_gemm<<<dim3(8, 16, 2), 128, 0, stream>>>(
      Gb, 0, YTk, sT2, NSEQ, DMODEL, NFPAD, NFPAD, NFPAD, nullptr,
      Kband, sRM, DMODEL, nullptr, 0, 0, 0, 0, nullptr, 0, 0);
  k_wmma_gemm<<<dim3(8, 16, 2), 128, 0, stream>>>(
      Gb, 0, YTv, sT2, NSEQ, DMODEL, NFPAD, NFPAD, NFPAD, nullptr,
      nullptr, 0, 0, VbT, (size_t)DMODEL * NSEQ, NSEQ, NSEQ, 0, nullptr, 0, 0);

  // 4) flash attention (time branch)
  k_flash_attn<<<dim3(NSEQ / 16, NHEADS, BATCH), 32, 0, stream>>>(Qband, Kband, VbT, timeO);

  // 5) frequency branch
  k_corr_reduce<<<dim3(NFREQ, BATCH), 256, 0, stream>>>(YTq, YTk, cr, ci);
  k_rmean<<<(BATCH * NSEQ) / 256, 256, 0, stream>>>(cr, ci, Rmean);
  k_topk<<<BATCH, 256, 0, stream>>>(Rmean, wbuf, taub);
  k_gather<<<BATCH * NSEQ, 256, 0, stream>>>(Vf, wbuf, taub, freqO);

  // 6) combine + output projection
  k_combine<<<(nX + 255) / 256, 256, 0, stream>>>(timeO, freqO, Hc, nX);
  k_wmma_gemm<<<dim3(8, 32, 1), 128, 0, stream>>>(
      Hc, 0, Wob, 0, BN, DMODEL, DMODEL, DMODEL, DMODEL, bo,
      nullptr, 0, 0, nullptr, 0, 0, 0, 0, Out, 0, DMODEL);
}
